// PixArtBlock_82265803587863
// MI455X (gfx1250) — matrix operand order repaired and checked by reading, not compiled, not run on hardware
//
#include <hip/hip_runtime.h>
#include <cstdint>
#include <cstddef>

// ---------------------------------------------------------------------------
// Types for CDNA5 WMMA (wave32): V_WMMA_F32_16X16X32_BF16
// ---------------------------------------------------------------------------
typedef __bf16 bf16;
typedef bf16  v16bf __attribute__((ext_vector_type(16)));
typedef float v8f   __attribute__((ext_vector_type(8)));

struct Halves { int4 lo; int4 hi; };
union FragB16 { v16bf v; Halves h; };

// Problem constants
#define CB   1152
#define NB   2048
#define HB   16
#define DHB  72
#define DHP  96      // head dim padded to 3*32 for WMMA K-loop

// ---------------------------------------------------------------------------
// f32 -> bf16 conversion (weights / activations)
// ---------------------------------------------------------------------------
__global__ void f32_to_bf16(const float* __restrict__ src, bf16* __restrict__ dst, long n) {
    long i = (long)blockIdx.x * 256 + threadIdx.x;
    if (i < n) dst[i] = (bf16)src[i];
}

// mods[b,i,c] = scale_shift_table[i,c] + t[b, i*C + c]
__global__ void build_mods(const float* __restrict__ sst, const float* __restrict__ t,
                           float* __restrict__ mods) {
    int i = blockIdx.x * 256 + threadIdx.x;
    if (i < 2 * 6 * CB) {
        int b = i / (6 * CB);
        int rc = i % (6 * CB);
        mods[i] = sst[rc] + t[b * 6 * CB + rc];
    }
}

// y [2,300,C] f32 -> yb [2,320,C] bf16 (zero padded rows)
__global__ void y_pad(const float* __restrict__ y, bf16* __restrict__ yb) {
    int s = blockIdx.x, b = blockIdx.y, tid = threadIdx.x;
#pragma unroll
    for (int j = 0; j < 5; ++j) {
        int c = tid + j * 256;
        if (c < CB) {
            float v = (s < 300) ? y[((long)b * 300 + s) * CB + c] : 0.f;
            yb[((long)b * 320 + s) * CB + c] = (bf16)v;
        }
    }
}

// ---------------------------------------------------------------------------
// Router: sigmoid(x @ router_w), rank-based dual top-k with sorted indices,
// per-bin weights and inverse maps.  One block per batch.
// ---------------------------------------------------------------------------
__global__ __launch_bounds__(256) void router_topk(
    const float* __restrict__ x, const float* __restrict__ rw,
    int* __restrict__ sidxL, float* __restrict__ wL, int* __restrict__ invL, int MpL, int KL,
    int* __restrict__ sidxU, float* __restrict__ wU, int* __restrict__ invU, int MpU, int KU)
{
    int b = blockIdx.x, tid = threadIdx.x;
    __shared__ float tw[NB];
    __shared__ int   rk[NB];
    for (int n = tid; n < NB; n += 256) {
        const float* xr = x + ((long)b * NB + n) * CB;
        float s = 0.f;
        for (int c = 0; c < CB; ++c) s += xr[c] * rw[c];
        tw[n] = 1.f / (1.f + __expf(-s));
    }
    __syncthreads();
    for (int n = tid; n < NB; n += 256) {
        float v = tw[n];
        int rr = 0;
        for (int m = 0; m < NB; ++m) {
            float u = tw[m];
            rr += (u > v) || (u == v && m < n);   // jax top_k tie-break: lower index first
        }
        rk[n] = rr;
    }
    __syncthreads();
    for (int n = tid; n < NB; n += 256) {
        int rr = rk[n];
        if (rr < KL) {
            int p = 0;
            for (int m = 0; m < n; ++m) p += (rk[m] < KL);
            sidxL[b * MpL + p] = n; wL[b * MpL + p] = tw[n]; invL[b * NB + n] = p;
        } else invL[b * NB + n] = -1;
        if (rr < KU) {
            int p = 0;
            for (int m = 0; m < n; ++m) p += (rk[m] < KU);
            sidxU[b * MpU + p] = n; wU[b * MpU + p] = tw[n]; invU[b * NB + n] = p;
        } else invU[b * NB + n] = -1;
    }
    for (int i = KL + tid; i < MpL; i += 256) { sidxL[b * MpL + i] = 0; wL[b * MpL + i] = 0.f; }
    for (int i = KU + tid; i < MpU; i += 256) { sidxU[b * MpU + i] = 0; wU[b * MpU + i] = 0.f; }
}

// ---------------------------------------------------------------------------
// (optional gather) + LayerNorm + t2i modulation -> bf16; optionally keeps raw
// f32 copy. Padded rows (i >= K) written as zeros.  grid(Mp, B), block 256.
// ---------------------------------------------------------------------------
__global__ __launch_bounds__(256) void gather_ln_mod(
    const float* __restrict__ src, const int* __restrict__ sidx, int K,
    int srcLd, long srcBatch, const float* __restrict__ mods,
    int shiftRow, int scaleRow,
    float* __restrict__ rawOut, bf16* __restrict__ hOut, int Mp)
{
    int i = blockIdx.x, b = blockIdx.y, tid = threadIdx.x;
    long orow = ((long)b * Mp + i) * CB;
    if (i >= K) {
        for (int c = tid; c < CB; c += 256) {
            if (rawOut) rawOut[orow + c] = 0.f;
            hOut[orow + c] = (bf16)0.f;
        }
        return;
    }
    int srcRow = sidx ? sidx[b * Mp + i] : i;
    const float* xr = src + (long)b * srcBatch + (long)srcRow * srcLd;
    float vals[5];
    float s1 = 0.f, s2 = 0.f;
#pragma unroll
    for (int j = 0; j < 5; ++j) {
        int c = tid + j * 256;
        float v = (c < CB) ? xr[c] : 0.f;
        vals[j] = v; s1 += v; s2 += v * v;
    }
    __shared__ float r1[256], r2[256];
    r1[tid] = s1; r2[tid] = s2;
    __syncthreads();
    for (int off = 128; off > 0; off >>= 1) {
        if (tid < off) { r1[tid] += r1[tid + off]; r2[tid] += r2[tid + off]; }
        __syncthreads();
    }
    float mu = r1[0] * (1.f / CB);
    float var = r2[0] * (1.f / CB) - mu * mu;
    float rs = rsqrtf(var + 1e-6f);
    const float* sh = mods + ((long)b * 6 + shiftRow) * CB;
    const float* sc = mods + ((long)b * 6 + scaleRow) * CB;
#pragma unroll
    for (int j = 0; j < 5; ++j) {
        int c = tid + j * 256;
        if (c < CB) {
            float v = vals[j];
            float hv = (v - mu) * rs * (1.f + sc[c]) + sh[c];
            if (rawOut) rawOut[orow + c] = v;
            hOut[orow + c] = (bf16)hv;
        }
    }
}

// ---------------------------------------------------------------------------
// Repack [*, H*72] slices into padded head layouts:
//   Qp/Kp: [B,H,S,96] (row-major d), Vt: [B,H,96,Sp] (row-major s)
// ---------------------------------------------------------------------------
__global__ void repack_q(const bf16* __restrict__ src, int ld, int off,
                         bf16* __restrict__ Qp, int Mp) {
    int m = blockIdx.x, h = blockIdx.y, b = blockIdx.z, d = threadIdx.x;
    bf16 v = (bf16)0.f;
    if (d < DHB) v = src[((long)b * Mp + m) * ld + off + h * DHB + d];
    Qp[((long)(b * HB + h) * Mp + m) * DHP + d] = v;
}

__global__ void repack_kv(const bf16* __restrict__ src, int ld, int offK, int offV,
                          bf16* __restrict__ Kp, bf16* __restrict__ Vt, int Sp) {
    int s = blockIdx.x, h = blockIdx.y, b = blockIdx.z, d = threadIdx.x;
    bf16 kv = (bf16)0.f, vv = (bf16)0.f;
    if (d < DHB) {
        long base = ((long)b * Sp + s) * ld;
        kv = src[base + offK + h * DHB + d];
        vv = src[base + offV + h * DHB + d];
    }
    Kp[((long)(b * HB + h) * Sp + s) * DHP + d] = kv;
    Vt[((long)(b * HB + h) * DHP + d) * Sp + s] = vv;
}

// ---------------------------------------------------------------------------
// WMMA GEMM:  out = epilogue( A[M,K] (bf16) x W[N,K]^T (bf16) )
// One wave -> 16x32 tile. A-tile staged to LDS with async-to-LDS double
// buffering; B fragments loaded straight from global (K-contiguous weights).
// Epilogue: +bias, optional tanh-GELU, per-column gate, per-row scale,
// +residual; writes f32 and/or bf16.
// ---------------------------------------------------------------------------
__global__ __launch_bounds__(256) void wmma_gemm(
    const bf16* __restrict__ A, int lda, long aBatch,
    const bf16* __restrict__ W,
    const float* __restrict__ bias,
    const float* __restrict__ colgate, long cgBatch,
    const float* __restrict__ rowscale, long rsBatch,
    const float* __restrict__ residual, long resBatch,
    float* __restrict__ outF, long ofBatch,
    bf16* __restrict__ outH, long ohBatch,
    int M, int Nc, int Kd, int act)
{
    const int lane = threadIdx.x;
    const int w    = threadIdx.y;
    const int b    = blockIdx.z;
    const int m0   = (blockIdx.y * 8 + w) * 16;
    const int n0   = blockIdx.x * 32;
    if (m0 >= M) return;                         // uniform per wave

    const int r  = lane & 15;
    const int hi = lane >> 4;

    __shared__ bf16 lds[8][2][512];              // 8 waves x 2 buffers x 1KB
    const unsigned ldsB0 = (unsigned)(size_t)&lds[w][0][0];
    const unsigned ldsB1 = (unsigned)(size_t)&lds[w][1][0];

    const bf16* Ab = A + (long)b * aBatch + (long)m0 * lda;
    const int o0 = lane * 16;                    // byte offset in tile, lanes cover 512B
    const int o1 = 512 + lane * 16;              // second half of 1KB tile
    const int row0 = o0 >> 6, col0 = (o0 & 63) >> 1;
    const int row1 = o1 >> 6, col1 = (o1 & 63) >> 1;

    v8f acc0 = {0.f,0.f,0.f,0.f,0.f,0.f,0.f,0.f};
    v8f acc1 = acc0;

    const int steps = Kd >> 5;

    // prologue: async-stage tile 0
    {
        unsigned long long g0 = (unsigned long long)(Ab + (long)row0 * lda + col0);
        unsigned long long g1 = (unsigned long long)(Ab + (long)row1 * lda + col1);
        asm volatile("global_load_async_to_lds_b128 %0, %1, off" :: "v"(ldsB0 + o0), "v"(g0) : "memory");
        asm volatile("global_load_async_to_lds_b128 %0, %1, off" :: "v"(ldsB0 + o1), "v"(g1) : "memory");
    }

    for (int t = 0; t < steps; ++t) {
        const int cur = t & 1;
        if (t + 1 < steps) {                     // software pipeline: prefetch t+1
            unsigned lb = cur ? ldsB0 : ldsB1;
            const int kk2 = (t + 1) * 32;
            unsigned long long g0 = (unsigned long long)(Ab + (long)row0 * lda + kk2 + col0);
            unsigned long long g1 = (unsigned long long)(Ab + (long)row1 * lda + kk2 + col1);
            asm volatile("global_load_async_to_lds_b128 %0, %1, off" :: "v"(lb + o0), "v"(g0) : "memory");
            asm volatile("global_load_async_to_lds_b128 %0, %1, off" :: "v"(lb + o1), "v"(g1) : "memory");
            asm volatile("s_wait_asynccnt 2" ::: "memory");  // tile t landed (in-order)
        } else {
            asm volatile("s_wait_asynccnt 0" ::: "memory");
        }

        // A fragment from LDS (contiguous k = 16*hi + s, same order as B)
        const bf16* lp = &lds[w][cur][0];
        FragB16 a;
        a.h.lo = *(const int4*)(lp + r * 32 + hi * 16);
        a.h.hi = *(const int4*)(lp + r * 32 + hi * 16 + 8);

        // B fragments straight from global (W is [Nc,Kd], K contiguous)
        const int kk = t * 32;
        const bf16* w0 = W + (long)(n0 + r) * Kd + kk + hi * 16;
        const bf16* w1 = W + (long)(n0 + 16 + r) * Kd + kk + hi * 16;
        FragB16 fb0, fb1;
        fb0.h.lo = *(const int4*)w0;       fb0.h.hi = *(const int4*)(w0 + 8);
        fb1.h.lo = *(const int4*)w1;       fb1.h.hi = *(const int4*)(w1 + 8);
        if (t + 1 < steps) __builtin_prefetch(w0 + 32, 0, 0);

        acc0 = __builtin_amdgcn_wmma_f32_16x16x32_bf16(false, a.v, false, fb0.v, (short)0, acc0, false, false);
        acc1 = __builtin_amdgcn_wmma_f32_16x16x32_bf16(false, a.v, false, fb1.v, (short)0, acc1, false, false);
    }

    // epilogue
    const float* cg  = colgate  ? colgate  + (long)b * cgBatch  : nullptr;
    const float* rs  = rowscale ? rowscale + (long)b * rsBatch  : nullptr;
    const float* res = residual ? residual + (long)b * resBatch : nullptr;
    float* oF = outF ? outF + (long)b * ofBatch : nullptr;
    bf16*  oH = outH ? outH + (long)b * ohBatch : nullptr;

#pragma unroll
    for (int half = 0; half < 2; ++half) {
        v8f acc = half ? acc1 : acc0;
        int n = n0 + half * 16 + r;
        float gate = cg ? cg[n] : 1.0f;
        float bv = bias ? bias[n] : 0.0f;
#pragma unroll
        for (int j = 0; j < 8; ++j) {
            int m = m0 + j + 8 * hi;
            float v = acc[j] + bv;
            if (act == 1) {                       // tanh GELU
                float u = v;
                float cc = 0.7978845608028654f * (u + 0.044715f * u * u * u);
                v = 0.5f * u * (1.0f + tanhf(cc));
            }
            float scaled = gate * (rs ? rs[m] : 1.0f) * v;
            float outv = (res ? res[(long)m * Nc + n] : 0.0f) + scaled;
            if (oF) oF[(long)m * Nc + n] = outv;
            if (oH) oH[(long)m * Nc + n] = (bf16)outv;
        }
    }
}

// ---------------------------------------------------------------------------
// Flash attention, WMMA scores + WMMA P*V.  One wave per (b, h, 16 q-rows).
// Online softmax in f32; score tiles transit LDS for the D->A transpose.
// grid(Mq/16, H, B), block 32.
// ---------------------------------------------------------------------------
__global__ __launch_bounds__(32) void attn_flash_wmma(
    const bf16* __restrict__ Qp, const bf16* __restrict__ Kp, const bf16* __restrict__ Vt,
    bf16* __restrict__ Out, int Mq, int Sp, int Sreal, float scale)
{
    const int lane = threadIdx.x;
    const int r = lane & 15, hi = lane >> 4;
    const int q0 = blockIdx.x * 16;
    const int h  = blockIdx.y;
    const int b  = blockIdx.z;

    const bf16* Qb = Qp + (long)(b * HB + h) * Mq * DHP;
    const bf16* Kb = Kp + (long)(b * HB + h) * Sp * DHP;
    const bf16* Vb = Vt + (long)(b * HB + h) * DHP * Sp;

    FragB16 qf[3];
#pragma unroll
    for (int t = 0; t < 3; ++t) {
        const bf16* qp = Qb + (long)(q0 + r) * DHP + t * 32;
        qf[t].h.lo = *(const int4*)(qp + hi * 16);
        qf[t].h.hi = *(const int4*)(qp + hi * 16 + 8);
    }

    v8f zero8 = {0.f,0.f,0.f,0.f,0.f,0.f,0.f,0.f};
    v8f acc[6];
#pragma unroll
    for (int t = 0; t < 6; ++t) acc[t] = zero8;
    float mr = -3.0e38f, lr = 0.f;

    __shared__ float Sbuf[16 * 32];
    __shared__ bf16  Pbuf[16 * 32];
    __shared__ float alphaB[16];
    __shared__ float lB[16];

    for (int s0 = 0; s0 < Sp; s0 += 32) {
        v8f d0 = zero8, d1 = zero8;
#pragma unroll
        for (int t = 0; t < 3; ++t) {
            const bf16* kp0 = Kb + (long)(s0 + r) * DHP + t * 32 + hi * 16;
            const bf16* kp1 = Kb + (long)(s0 + 16 + r) * DHP + t * 32 + hi * 16;
            FragB16 bk0, bk1;
            bk0.h.lo = *(const int4*)kp0; bk0.h.hi = *(const int4*)(kp0 + 8);
            bk1.h.lo = *(const int4*)kp1; bk1.h.hi = *(const int4*)(kp1 + 8);
            d0 = __builtin_amdgcn_wmma_f32_16x16x32_bf16(false, qf[t].v, false, bk0.v, (short)0, d0, false, false);
            d1 = __builtin_amdgcn_wmma_f32_16x16x32_bf16(false, qf[t].v, false, bk1.v, (short)0, d1, false, false);
        }
        // scatter D (lane=col) into row-major Sbuf
#pragma unroll
        for (int j = 0; j < 8; ++j) {
            Sbuf[(j + 8 * hi) * 32 + r]      = d0[j] * scale;
            Sbuf[(j + 8 * hi) * 32 + 16 + r] = d1[j] * scale;
        }
        asm volatile("s_wait_dscnt 0" ::: "memory");

        // online softmax, lane handles row (lane&15); hi-lanes duplicate (benign)
        const float* srow = &Sbuf[r * 32];
        float rowv[32];
        float mx = -3.0e38f;
#pragma unroll
        for (int c = 0; c < 32; ++c) {
            float v = srow[c];
            v = (s0 + c < Sreal) ? v : -3.0e38f;
            rowv[c] = v;
            mx = fmaxf(mx, v);
        }
        float mnew = fmaxf(mr, mx);
        float al = __expf(mr - mnew);
        float ssum = 0.f;
#pragma unroll
        for (int c = 0; c < 32; ++c) {
            float p = __expf(rowv[c] - mnew);
            ssum += p;
            Pbuf[r * 32 + c] = (bf16)p;
        }
        lr = lr * al + ssum;
        mr = mnew;
        alphaB[r] = al;
        asm volatile("s_wait_dscnt 0" ::: "memory");

        // rescale accumulators (per-row alpha)
        float av[8];
#pragma unroll
        for (int j = 0; j < 8; ++j) av[j] = alphaB[j + 8 * hi];
#pragma unroll
        for (int t = 0; t < 6; ++t)
#pragma unroll
            for (int j = 0; j < 8; ++j) acc[t][j] *= av[j];

        // P * V
        FragB16 pf;
        const bf16* pp = &Pbuf[r * 32];
        pf.h.lo = *(const int4*)(pp + hi * 16);
        pf.h.hi = *(const int4*)(pp + hi * 16 + 8);
#pragma unroll
        for (int t = 0; t < 6; ++t) {
            const bf16* vp = Vb + (long)(t * 16 + r) * Sp + s0 + hi * 16;
            FragB16 bv;
            bv.h.lo = *(const int4*)vp; bv.h.hi = *(const int4*)(vp + 8);
            acc[t] = __builtin_amdgcn_wmma_f32_16x16x32_bf16(false, pf.v, false, bv.v, (short)0, acc[t], false, false);
        }
    }

    lB[r] = lr;
    asm volatile("s_wait_dscnt 0" ::: "memory");
    float lv[8];
#pragma unroll
    for (int j = 0; j < 8; ++j) lv[j] = lB[j + 8 * hi];

#pragma unroll
    for (int t = 0; t < 6; ++t) {
        int d = t * 16 + r;
        if (d < DHB) {
#pragma unroll
            for (int j = 0; j < 8; ++j) {
                int q = q0 + j + 8 * hi;
                Out[((long)b * Mq + q) * CB + h * DHB + d] = (bf16)(acc[t][j] / lv[j]);
            }
        }
    }
}

// ---------------------------------------------------------------------------
// Final blend:  out = lw * bin_lower_full + uw * bin_upper_full
// ---------------------------------------------------------------------------
__global__ void final_combine(const float* __restrict__ x, const float* __restrict__ dr,
                              const int* __restrict__ invL, const float* __restrict__ binL, int MpL,
                              const int* __restrict__ invU, const float* __restrict__ binU, int MpU,
                              float* __restrict__ out)
{
    int n = blockIdx.x, b = blockIdx.y, tid = threadIdx.x;
    float kr = fminf(fmaxf(dr[0], 0.1f), 1.0f);
    float lw = (0.8f - kr) * 10.f;     // (UPPER-kr)/(UPPER-LOWER)
    float uw = 1.f - lw;
    int pl = invL[b * NB + n];
    int pu = invU[b * NB + n];
#pragma unroll
    for (int j = 0; j < 5; ++j) {
        int c = tid + j * 256;
        if (c < CB) {
            float xv = x[((long)b * NB + n) * CB + c];
            float vl = (pl >= 0) ? binL[((long)b * MpL + pl) * CB + c] : xv;
            float vu = (pu >= 0) ? binU[((long)b * MpU + pu) * CB + c] : xv;
            out[((long)b * NB + n) * CB + c] = lw * vl + uw * vu;
        }
    }
}

// ---------------------------------------------------------------------------
// Host orchestration
// ---------------------------------------------------------------------------
extern "C" void kernel_launch(void* const* d_in, const int* in_sizes, int n_in,
                              void* d_out, int out_size, void* d_ws, size_t ws_size,
                              hipStream_t stream) {
    (void)in_sizes; (void)n_in; (void)out_size; (void)ws_size;
    const int B = 2, KL = 1434, KU = 1639;
    const int MpL = 1440, MpU = 1664;          // pad32
    const int SY = 300, SpY = 320;

    const float* x_f   = (const float*)d_in[0];
    const float* y_f   = (const float*)d_in[1];
    const float* t_f   = (const float*)d_in[2];
    const float* dr_f  = (const float*)d_in[3];
    const float* rw_f  = (const float*)d_in[4];
    const float* sst_f = (const float*)d_in[5];

    // workspace carving
    size_t off = 0; char* base = (char*)d_ws;
    auto carve = [&](size_t bytes) -> void* {
        void* p = base + off;
        off = (off + bytes + 255) & ~(size_t)255;
        return p;
    };
    bf16* qkv_wb    = (bf16*)carve((size_t)3456 * CB * 2);
    bf16* aproj_wb  = (bf16*)carve((size_t)CB * CB * 2);
    bf16* q_wb      = (bf16*)carve((size_t)CB * CB * 2);
    bf16* kv_wb     = (bf16*)carve((size_t)2304 * CB * 2);
    bf16* caproj_wb = (bf16*)carve((size_t)CB * CB * 2);
    bf16* fc1_wb    = (bf16*)carve((size_t)4608 * CB * 2);
    bf16* fc2_wb    = (bf16*)carve((size_t)CB * 4608 * 2);
    float* mods     = (float*)carve((size_t)B * 6 * CB * 4);
    bf16* yb        = (bf16*)carve((size_t)B * SpY * CB * 2);
    bf16* kvb       = (bf16*)carve((size_t)B * SpY * 2304 * 2);
    bf16* KpY       = (bf16*)carve((size_t)B * HB * SpY * DHP * 2);
    bf16* VtY       = (bf16*)carve((size_t)B * HB * DHP * SpY * 2);
    int*   sidxL    = (int*)carve((size_t)B * MpL * 4);
    float* wsL      = (float*)carve((size_t)B * MpL * 4);
    int*   invL     = (int*)carve((size_t)B * NB * 4);
    int*   sidxU    = (int*)carve((size_t)B * MpU * 4);
    float* wsU      = (float*)carve((size_t)B * MpU * 4);
    int*   invU     = (int*)carve((size_t)B * NB * 4);
    float* binL     = (float*)carve((size_t)B * MpL * CB * 4);
    float* binU     = (float*)carve((size_t)B * MpU * CB * 4);
    // shared per-bin scratch (sized for the larger bin)
    float* tkx      = (float*)carve((size_t)B * MpU * CB * 4);
    bf16* txb       = (bf16*)carve((size_t)B * MpU * CB * 2);
    bf16* hb        = (bf16*)carve((size_t)B * MpU * CB * 2);
    bf16* qkvb      = (bf16*)carve((size_t)B * MpU * 3456 * 2);
    bf16* Qp        = (bf16*)carve((size_t)B * HB * MpU * DHP * 2);
    bf16* Kp        = (bf16*)carve((size_t)B * HB * MpU * DHP * 2);
    bf16* Vt        = (bf16*)carve((size_t)B * HB * DHP * MpU * 2);
    bf16* attn_o    = (bf16*)carve((size_t)B * MpU * CB * 2);
    bf16* qb        = (bf16*)carve((size_t)B * MpU * CB * 2);
    bf16* m1        = (bf16*)carve((size_t)B * MpU * 4608 * 2);

    auto cvt = [&](const void* src, bf16* dst, long n) {
        f32_to_bf16<<<dim3((unsigned)((n + 255) / 256)), dim3(256), 0, stream>>>((const float*)src, dst, n);
    };
    cvt(d_in[6],  qkv_wb,    3456l * CB);
    cvt(d_in[8],  aproj_wb,  (long)CB * CB);
    cvt(d_in[10], q_wb,      (long)CB * CB);
    cvt(d_in[12], kv_wb,     2304l * CB);
    cvt(d_in[14], caproj_wb, (long)CB * CB);
    cvt(d_in[16], fc1_wb,    4608l * CB);
    cvt(d_in[18], fc2_wb,    (long)CB * 4608);

    build_mods<<<dim3((2 * 6 * CB + 255) / 256), dim3(256), 0, stream>>>(sst_f, t_f, mods);
    y_pad<<<dim3(SpY, B), dim3(256), 0, stream>>>(y_f, yb);
    router_topk<<<dim3(B), dim3(256), 0, stream>>>(x_f, rw_f,
        sidxL, wsL, invL, MpL, KL, sidxU, wsU, invU, MpU, KU);

    auto gemm = [&](const bf16* A, int lda, long aB, const bf16* W, const float* bias,
                    const float* cg, long cgB, const float* rs, long rsB,
                    const float* res, long resB, float* oF, long oFB, bf16* oH, long oHB,
                    int M, int Nc, int Kd, int act) {
        dim3 g(Nc / 32, (M + 127) / 128, B), blk(32, 8);
        wmma_gemm<<<g, blk, 0, stream>>>(A, lda, aB, W, bias, cg, cgB, rs, rsB,
                                         res, resB, oF, oFB, oH, oHB, M, Nc, Kd, act);
    };

    // bin-independent cross-attn K/V
    gemm(yb, CB, (long)SpY * CB, kv_wb, (const float*)d_in[13],
         nullptr, 0, nullptr, 0, nullptr, 0, nullptr, 0, kvb, (long)SpY * 2304,
         SpY, 2304, CB, 0);
    repack_kv<<<dim3(SpY, HB, B), dim3(DHP), 0, stream>>>(kvb, 2304, 0, 1152, KpY, VtY, SpY);

    const float kscale = 0.11785113019775793f;   // 72^-0.5

    auto run_bin = [&](int K, int Mp, const int* sidx, const float* wsort, float* binout) {
        gather_ln_mod<<<dim3(Mp, B), dim3(256), 0, stream>>>(
            x_f, sidx, K, CB, (long)NB * CB, mods, 0, 1, tkx, hb, Mp);
        gemm(hb, CB, (long)Mp * CB, qkv_wb, (const float*)d_in[7],
             nullptr, 0, nullptr, 0, nullptr, 0, nullptr, 0, qkvb, (long)Mp * 3456,
             Mp, 3456, CB, 0);
        repack_q <<<dim3(Mp, HB, B), dim3(DHP), 0, stream>>>(qkvb, 3456, 0, Qp, Mp);
        repack_kv<<<dim3(Mp, HB, B), dim3(DHP), 0, stream>>>(qkvb, 3456, 1152, 2304, Kp, Vt, Mp);
        attn_flash_wmma<<<dim3(Mp / 16, HB, B), dim3(32), 0, stream>>>(
            Qp, Kp, Vt, attn_o, Mp, Mp, K, kscale);
        // topk_x += gmsa * (attn_o @ Wp^T + b); mirror bf16
        gemm(attn_o, CB, (long)Mp * CB, aproj_wb, (const float*)d_in[9],
             mods + 2 * CB, (long)6 * CB, nullptr, 0,
             tkx, (long)Mp * CB, tkx, (long)Mp * CB, txb, (long)Mp * CB,
             Mp, CB, CB, 0);
        gemm(txb, CB, (long)Mp * CB, q_wb, (const float*)d_in[11],
             nullptr, 0, nullptr, 0, nullptr, 0, nullptr, 0, qb, (long)Mp * CB,
             Mp, CB, CB, 0);
        repack_q<<<dim3(Mp, HB, B), dim3(DHP), 0, stream>>>(qb, CB, 0, Qp, Mp);
        attn_flash_wmma<<<dim3(Mp / 16, HB, B), dim3(32), 0, stream>>>(
            Qp, KpY, VtY, attn_o, Mp, SpY, SY, kscale);
        // topk_x += ca @ Wp^T + b
        gemm(attn_o, CB, (long)Mp * CB, caproj_wb, (const float*)d_in[15],
             nullptr, 0, nullptr, 0,
             tkx, (long)Mp * CB, tkx, (long)Mp * CB, nullptr, 0,
             Mp, CB, CB, 0);
        gather_ln_mod<<<dim3(Mp, B), dim3(256), 0, stream>>>(
            tkx, nullptr, K, CB, (long)Mp * CB, mods, 3, 4, nullptr, hb, Mp);
        gemm(hb, CB, (long)Mp * CB, fc1_wb, (const float*)d_in[17],
             nullptr, 0, nullptr, 0, nullptr, 0, nullptr, 0, m1, (long)Mp * 4608,
             Mp, 4608, CB, 1);
        // binout = topk_x + w_sorted * gmlp * (m1 @ fc2^T + b)
        gemm(m1, 4608, (long)Mp * 4608, fc2_wb, (const float*)d_in[19],
             mods + 5 * CB, (long)6 * CB, wsort, (long)Mp,
             tkx, (long)Mp * CB, binout, (long)Mp * CB, nullptr, 0,
             Mp, CB, 4608, 0);
    };

    run_bin(KL, MpL, sidxL, wsL, binL);
    run_bin(KU, MpU, sidxU, wsU, binU);

    final_combine<<<dim3(NB, B), dim3(256), 0, stream>>>(
        x_f, dr_f, invL, binL, MpL, invU, binU, MpU, (float*)d_out);
}
